// CoordinateDecoder_32865089749206
// MI455X (gfx1250) — matrix operand order repaired and checked by reading, not compiled, not run on hardware
//
#include <hip/hip_runtime.h>
#include <hip/hip_bf16.h>
#include <math.h>

typedef __attribute__((ext_vector_type(16))) __bf16 v16bf;
typedef __attribute__((ext_vector_type(8)))  float  v8f;
typedef __attribute__((ext_vector_type(4)))  int    v4i;

#define NPTS   16384
#define NBATCH 8
#define MTOT   (NBATCH * NPTS)   // 131072
#define HIDDIM 256

#define AS1 __attribute__((address_space(1)))
#define AS3 __attribute__((address_space(3)))

#if defined(__gfx1250__) && __has_builtin(__builtin_amdgcn_global_load_async_to_lds_b128)
#define USE_ASYNC_LDS 1
#else
#define USE_ASYNC_LDS 0
#endif

// Branch-free fast sigmoid / tanh / GELU on hardware transcendentals.
// Limits are safe: __expf(+big)=inf, rcp(inf)=0.
static __device__ __forceinline__ float fast_sigmoid(float t) {
    return __builtin_amdgcn_rcpf(1.0f + __expf(-t));
}
static __device__ __forceinline__ float fast_tanh(float t) {
    return 2.0f * fast_sigmoid(2.0f * t) - 1.0f;
}
static __device__ __forceinline__ float gelu_tanh(float x) {
    // 0.5*x*(1+tanh(u)) == x * sigmoid(2u), u = 0.7978845608*(x + 0.044715 x^3)
    float u2 = 1.5957691216057308f * (x + 0.044715f * x * x * x);
    return x * fast_sigmoid(u2);
}

// ---------------- weight conversion (pad w0 rows 234..255 with zeros) ----------------
__global__ void convert_w_kernel(const float* __restrict__ src, __bf16* __restrict__ dst,
                                 int valid_rows) {
    int idx = blockIdx.x * blockDim.x + threadIdx.x;   // 0..65535
    int k = idx >> 8, n = idx & 255;
    float v = (k < valid_rows) ? src[k * 256 + n] : 0.0f;
    dst[idx] = (__bf16)v;
}

// ---------------- FiLM: gamma/beta = ctx @ film_w + film_b (+1 for gamma) ----------------
__global__ void film_kernel(const float* __restrict__ ctx, const float* __restrict__ fw,
                            const float* __restrict__ fb, float* __restrict__ gb) {
    int idx = blockIdx.x * blockDim.x + threadIdx.x;   // 0..4095
    int b = idx >> 9, j = idx & 511;
    float s = fb[j];
    for (int k = 0; k < 256; ++k) s += ctx[b * 256 + k] * fw[k * 512 + j];
    if (j < 256) s += 1.0f;
    gb[b * 512 + j] = s;
}

// ---------------- antialiased bilinear downsample (jax.image.resize semantics) ----------
static __device__ __forceinline__ int resize_taps(int i, int s, int Hi, float* w) {
    float center = (i + 0.5f) * (float)s - 0.5f;
    int lo = (int)ceilf(center - (float)s);
    int hi = (int)floorf(center + (float)s);
    if (lo < 0) lo = 0;
    if (hi > Hi - 1) hi = Hi - 1;
    float sum = 0.0f;
    int c = 0;
    for (int j = lo; j <= hi; ++j) {
        float wt = 1.0f - fabsf((float)j - center) / (float)s;
        if (wt <= 0.0f) { if (c == 0) lo++; continue; }
        w[c++] = wt;
        sum += wt;
    }
    for (int k = 0; k < c; ++k) w[k] /= sum;
    return (c << 16) | (lo & 0xFFFF);
}

#define L1ELEMS (NBATCH * 128 * 128 * 64)
#define L2ELEMS (NBATCH * 64 * 64 * 64)

__global__ void pyramid_kernel(const float* __restrict__ g, float* __restrict__ p1,
                               float* __restrict__ p2) {
    long long idx = (long long)blockIdx.x * blockDim.x + threadIdx.x;
    int s, Ho;
    float* dst;
    long long e;
    if (idx < L1ELEMS) { s = 2; Ho = 128; dst = p1; e = idx; }
    else if (idx < (long long)L1ELEMS + L2ELEMS) { s = 4; Ho = 64; dst = p2; e = idx - L1ELEMS; }
    else return;
    int c = (int)(e & 63);
    int x = (int)((e >> 6) % Ho);
    int y = (int)((e >> 6) / Ho) % Ho;
    int b = (int)(e / ((long long)64 * Ho * Ho));
    float wy[8], wx[8];
    int ty = resize_taps(y, s, 256, wy);
    int tx = resize_taps(x, s, 256, wx);
    int cy = ty >> 16, y0 = ty & 0xFFFF;
    int cx = tx >> 16, x0 = tx & 0xFFFF;
    const float* base = g + (size_t)b * 256 * 256 * 64;
    float acc = 0.0f;
    for (int jy = 0; jy < cy; ++jy) {
        const float* rowp = base + ((size_t)(y0 + jy) * 256) * 64;
        float a = 0.0f;
        for (int jx = 0; jx < cx; ++jx) a += wx[jx] * rowp[(size_t)(x0 + jx) * 64 + c];
        acc += wy[jy] * a;
    }
    dst[e] = acc;
}

// ---------------- bilinear sample helper (matches reference _sample_level) --------------
static __device__ __forceinline__ float bilin(const float* __restrict__ g, int b, int HW,
                                              float c0, float c1, int c) {
    float fy = (c0 + 1.0f) * 0.5f * (float)(HW - 1);
    float fx = (c1 + 1.0f) * 0.5f * (float)(HW - 1);
    float y0f = floorf(fy), x0f = floorf(fx);
    float wy = fy - y0f, wx = fx - x0f;
    int y0 = min(max((int)y0f, 0), HW - 1);
    int y1 = min(max((int)y0f + 1, 0), HW - 1);
    int x0 = min(max((int)x0f, 0), HW - 1);
    int x1 = min(max((int)x0f + 1, 0), HW - 1);
    const float* base = g + (size_t)b * HW * HW * 64;
    float g00 = base[((size_t)y0 * HW + x0) * 64 + c];
    float g01 = base[((size_t)y0 * HW + x1) * 64 + c];
    float g10 = base[((size_t)y1 * HW + x0) * 64 + c];
    float g11 = base[((size_t)y1 * HW + x1) * 64 + c];
    return g00 * (1 - wy) * (1 - wx) + g01 * (1 - wy) * wx +
           g10 * wy * (1 - wx) + g11 * wy * wx;
}

// ---------------- assemble per-point input row (256 cols, bf16) -------------------------
__global__ void assemble_kernel(const float* __restrict__ grid, const float* __restrict__ p1,
                                const float* __restrict__ p2, const float* __restrict__ coords,
                                __bf16* __restrict__ A0) {
    int p = blockIdx.x * 4 + (threadIdx.x >> 6);   // point 0..131071
    int c = threadIdx.x & 63;                      // channel 0..63
    int b = p >> 14;
    int n = p & 16383;
    float c0 = coords[n * 2 + 0];
    float c1 = coords[n * 2 + 1];
    size_t ro = (size_t)p * 256;
    A0[ro + 42 + c]  = (__bf16)bilin(grid, b, 256, c0, c1, c);
    A0[ro + 106 + c] = (__bf16)bilin(p1,   b, 128, c0, c1, c);
    A0[ro + 170 + c] = (__bf16)bilin(p2,   b, 64,  c0, c1, c);
    if (c < 42) {
        float v;
        if (c == 0) v = c0;
        else if (c == 1) v = c1;
        else {
            int e = c - 2;
            int fi = e >> 2, sc = (e >> 1) & 1, d = e & 1;
            float f = (float)(1 << fi) * 3.14159265358979323846f;
            float xd = d ? c1 : c0;
            v = sc ? __cosf(xd * f) : __sinf(xd * f);
        }
        A0[ro + c] = (__bf16)v;
    }
    if (c < 22) A0[ro + 234 + c] = (__bf16)0.0f;
}

// ---------------- fused GEMM + bias + FiLM + GELU (bf16 WMMA, f32 accum) ----------------
// A: M x 256 bf16 row-major.  W: 256 x 256 bf16 row-major (K x N).
// Block tile 128x128, 256 threads = 8 waves in 4(M) x 2(N), each wave 2x4 WMMA tiles.
#define XS_STRIDE 40   // 80B rows: 16B-aligned fragments, de-skewed banks
#define WS_STRIDE 40

__global__ void __launch_bounds__(256)
gemm_film_gelu_kernel(const __bf16* __restrict__ A, const __bf16* __restrict__ W,
                      const float* __restrict__ bias, const float* __restrict__ gb,
                      __bf16* __restrict__ out) {
    __shared__ __bf16 Xs[128][XS_STRIDE];   // X tile, row-major (cols 0..31 used)
    __shared__ __bf16 Ws[128][WS_STRIDE];   // W tile transposed: Ws[n][k]
    __shared__ __bf16 Ds[128][128];         // output staging for coalesced stores
    const int tid  = threadIdx.x;
    const int wv   = tid >> 5;
    const int lane = tid & 31;
    const int half = lane >> 4;
    const int r    = lane & 15;
    const int wm   = wv & 3;     // 0..3 -> 32-row slab
    const int wn   = wv >> 2;    // 0..1 -> 64-col slab
    const size_t m0 = (size_t)blockIdx.x * 128;
    const int    n0 = blockIdx.y * 128;

    v8f acc[2][4];
#pragma unroll
    for (int i = 0; i < 2; ++i)
#pragma unroll
        for (int j = 0; j < 4; ++j) acc[i][j] = (v8f){0, 0, 0, 0, 0, 0, 0, 0};

    for (int kc = 0; kc < 256; kc += 32) {
        // X tile: 4096 bf16, 16 per thread via two 16B transfers
#pragma unroll
        for (int i = 0; i < 2; ++i) {
            int idx = tid + i * 256;
            int row = idx >> 2, q = idx & 3;
#if USE_ASYNC_LDS
            __builtin_amdgcn_global_load_async_to_lds_b128(
                (AS1 v4i*)(A + (m0 + row) * 256 + kc + q * 8),
                (AS3 v4i*)&Xs[row][q * 8], 0, 0);
#else
            const uint4* src = reinterpret_cast<const uint4*>(A + (m0 + row) * 256 + kc + q * 8);
            *reinterpret_cast<uint4*>(&Xs[row][q * 8]) = *src;
#endif
        }
        // W tile transposed into LDS (coalesced global reads along N)
        {
            int n = tid & 127, kh = tid >> 7;
#pragma unroll
            for (int kk = 0; kk < 16; ++kk) {
                int k = kh * 16 + kk;
                Ws[n][k] = W[(size_t)(kc + k) * 256 + n0 + n];
            }
        }
#if USE_ASYNC_LDS
#if __has_builtin(__builtin_amdgcn_s_wait_asynccnt)
        __builtin_amdgcn_s_wait_asynccnt(0);
#else
        asm volatile("s_wait_asynccnt 0" ::: "memory");
#endif
#endif
        __syncthreads();

        // A fragment (ISA 16-bit A 16x32 layout): lane half selects K {0..7,16..23} vs {8..15,24..31}
        v16bf af[2], bfr[4];
#pragma unroll
        for (int i = 0; i < 2; ++i) {
            const __bf16* base = &Xs[wm * 32 + i * 16 + r][half * 8];
            union { v16bf v; float4 f[2]; } u;
            u.f[0] = *reinterpret_cast<const float4*>(base);
            u.f[1] = *reinterpret_cast<const float4*>(base + 16);
            af[i] = u.v;
        }
        // B fragment (same K order as A): lane half selects K {0..7,16..23} vs {8..15,24..31}
#pragma unroll
        for (int j = 0; j < 4; ++j) {
            const __bf16* base = &Ws[wn * 64 + j * 16 + r][half * 8];
            union { v16bf v; float4 f[2]; } u;
            u.f[0] = *reinterpret_cast<const float4*>(base);
            u.f[1] = *reinterpret_cast<const float4*>(base + 16);
            bfr[j] = u.v;
        }
#pragma unroll
        for (int i = 0; i < 2; ++i)
#pragma unroll
            for (int j = 0; j < 4; ++j)
                acc[i][j] = __builtin_amdgcn_wmma_f32_16x16x32_bf16(
                    false, af[i], false, bfr[j], (short)0, acc[i][j], false, false);
        __syncthreads();
    }

    // Epilogue: bias + FiLM + GELU into LDS staging (D layout: VGPR g -> row g+8*half)
    const int batch = (int)(m0 >> 14);   // whole 128-row tile lies in one batch
#pragma unroll
    for (int i = 0; i < 2; ++i) {
#pragma unroll
        for (int j = 0; j < 4; ++j) {
            int lcol = wn * 64 + j * 16 + r;
            int gcol = n0 + lcol;
            float bv = bias[gcol];
            float gm = gb[batch * 512 + gcol];
            float bt = gb[batch * 512 + 256 + gcol];
            int lrowb = wm * 32 + i * 16 + half * 8;
#pragma unroll
            for (int g = 0; g < 8; ++g) {
                float v = acc[i][j][g] + bv;
                v = gelu_tanh(v * gm + bt);
                Ds[lrowb + g][lcol] = (__bf16)v;
            }
        }
    }
    __syncthreads();
    // Coalesced flush: 2048 16B chunks, 8 per thread
#pragma unroll
    for (int t = 0; t < 8; ++t) {
        int chunk = tid + t * 256;
        int row = chunk >> 4, q = chunk & 15;
        *reinterpret_cast<uint4*>(out + (m0 + row) * 256 + n0 + q * 8) =
            *reinterpret_cast<const uint4*>(&Ds[row][q * 8]);
    }
}

// ---------------- output head: 256 -> 4, tanh/sigmoid -----------------------------------
__global__ void out_kernel(const __bf16* __restrict__ act, const float* __restrict__ wout,
                           const float* __restrict__ bout, float* __restrict__ out) {
    int m = blockIdx.x * blockDim.x + threadIdx.x;
    float s0 = bout[0], s1 = bout[1], s2 = bout[2], s3 = bout[3];
    const __bf16* row = act + (size_t)m * 256;
    for (int k = 0; k < 256; ++k) {
        float a = (float)row[k];
        const float* wr = wout + k * 4;
        s0 += a * wr[0]; s1 += a * wr[1]; s2 += a * wr[2]; s3 += a * wr[3];
    }
    float* o = out + (size_t)m * 4;
    o[0] = fast_tanh(s0);
    o[1] = fast_tanh(s1);
    o[2] = fast_tanh(s2);
    o[3] = fast_sigmoid(s3);
}

extern "C" void kernel_launch(void* const* d_in, const int* in_sizes, int n_in,
                              void* d_out, int out_size, void* d_ws, size_t ws_size,
                              hipStream_t stream) {
    const float* feature_grid = (const float*)d_in[0];
    const float* ctx          = (const float*)d_in[1];
    const float* coords       = (const float*)d_in[2];
    const float* w0 = (const float*)d_in[3];  const float* b0 = (const float*)d_in[4];
    const float* w1 = (const float*)d_in[5];  const float* b1 = (const float*)d_in[6];
    const float* w2 = (const float*)d_in[7];  const float* b2 = (const float*)d_in[8];
    const float* w3 = (const float*)d_in[9];  const float* b3 = (const float*)d_in[10];
    const float* film_w = (const float*)d_in[11];
    const float* film_b = (const float*)d_in[12];
    const float* wout   = (const float*)d_in[13];
    const float* bout   = (const float*)d_in[14];

    char* ws = (char*)d_ws;
    size_t off = 0;
    float* gb = (float*)(ws + off); off += (size_t)NBATCH * 512 * sizeof(float);
    float* p1 = (float*)(ws + off); off += (size_t)L1ELEMS * sizeof(float);
    float* p2 = (float*)(ws + off); off += (size_t)L2ELEMS * sizeof(float);
    __bf16* wb0 = (__bf16*)(ws + off); off += 65536 * sizeof(__bf16);
    __bf16* wb1 = (__bf16*)(ws + off); off += 65536 * sizeof(__bf16);
    __bf16* wb2 = (__bf16*)(ws + off); off += 65536 * sizeof(__bf16);
    __bf16* wb3 = (__bf16*)(ws + off); off += 65536 * sizeof(__bf16);
    __bf16* actA = (__bf16*)(ws + off); off += (size_t)MTOT * 256 * sizeof(__bf16);
    __bf16* actB = (__bf16*)(ws + off); off += (size_t)MTOT * 256 * sizeof(__bf16);

    convert_w_kernel<<<256, 256, 0, stream>>>(w0, wb0, 234);
    convert_w_kernel<<<256, 256, 0, stream>>>(w1, wb1, 256);
    convert_w_kernel<<<256, 256, 0, stream>>>(w2, wb2, 256);
    convert_w_kernel<<<256, 256, 0, stream>>>(w3, wb3, 256);
    film_kernel<<<16, 256, 0, stream>>>(ctx, film_w, film_b, gb);

    long long pyr_total = (long long)L1ELEMS + L2ELEMS;
    pyramid_kernel<<<(unsigned)((pyr_total + 255) / 256), 256, 0, stream>>>(feature_grid, p1, p2);

    assemble_kernel<<<MTOT / 4, 256, 0, stream>>>(feature_grid, p1, p2, coords, actA);

    dim3 ggrid(MTOT / 128, 2);
    gemm_film_gelu_kernel<<<ggrid, 256, 0, stream>>>(actA, wb0, b0, gb, actB);
    gemm_film_gelu_kernel<<<ggrid, 256, 0, stream>>>(actB, wb1, b1, gb, actA);
    gemm_film_gelu_kernel<<<ggrid, 256, 0, stream>>>(actA, wb2, b2, gb, actB);
    gemm_film_gelu_kernel<<<ggrid, 256, 0, stream>>>(actB, wb3, b3, gb, actA);

    out_kernel<<<MTOT / 256, 256, 0, stream>>>(actA, wout, bout, (float*)d_out);
}
